// XlaQuantizedLinear_68307159875942
// MI455X (gfx1250) — matrix operand order repaired and checked by reading, not compiled, not run on hardware
//
#include <hip/hip_runtime.h>

typedef __attribute__((ext_vector_type(16))) _Float16 v16h;
typedef __attribute__((ext_vector_type(8)))  _Float16 v8h;
typedef __attribute__((ext_vector_type(8)))  float    v8f;
typedef __attribute__((ext_vector_type(4)))  float    f32x4;
typedef __attribute__((ext_vector_type(4)))  int      i32x4;

namespace {
constexpr int M_TOT  = 2 * 2048;     // B * S
constexpr int K_TOT  = 4096;         // IN_DIM
constexpr int N_TOT  = 16384;        // OUT_DIM
constexpr int BLK_M  = 256;
constexpr int BLK_N  = 128;
constexpr int KT     = 64;           // K per LDS stage = 2 WMMA k-steps
constexpr int LDH    = 72;           // padded LDS row stride in halves (16B-aligned)
constexpr int NSTAGE = K_TOT / KT;   // 64
}

__global__ __launch_bounds__(256)
void qlin_wmma_f16(const float* __restrict__ X,
                   const int*   __restrict__ W,
                   const float* __restrict__ SC,
                   float*       __restrict__ O)
{
    __shared__ alignas(128) _Float16 lA[BLK_M * LDH];   // x tile  [256 rows][64 k]  36864 B
    __shared__ alignas(128) _Float16 lB[BLK_N * LDH];   // w tile  [128 cols][64 k]  18432 B

    const int t    = threadIdx.x;
    const int lane = t & 31;
    const int g    = lane >> 4;      // half-wave group (0/1)
    const int lr   = lane & 15;
    const int wid  = t >> 5;
    const int wm   = wid >> 1;       // 0..3  -> M offset wm*64
    const int wn   = wid & 1;        // 0..1  -> N offset wn*64

    const int m0 = blockIdx.x * BLK_M;   // M-fast grid order => weight-stripe L2 reuse
    const int n0 = blockIdx.y * BLK_N;

    // Staging map:
    //  A: each thread owns one full 64-wide row:   row = t        (256 rows)
    //  B: each thread owns a 32-wide half row:     row = t>>1     (128 rows)
    const int rowB = t >> 1;
    const int colB = (t & 1) * 32;

    const float* gA = X + (size_t)(m0 + t) * K_TOT;
    const int*   gB = W + (size_t)(n0 + rowB) * K_TOT + colB;

    f32x4 ra[16];
    i32x4 rb[8];

    v8f acc[4][4];
    #pragma unroll
    for (int i = 0; i < 4; ++i)
        #pragma unroll
        for (int j = 0; j < 4; ++j)
            #pragma unroll
            for (int e = 0; e < 8; ++e)
                acc[i][j][e] = 0.0f;

    auto loadStage = [&](int kt) {
        const f32x4* pa = (const f32x4*)(gA + (size_t)kt * KT);
        const i32x4* pb = (const i32x4*)(gB + (size_t)kt * KT);
        #pragma unroll
        for (int i = 0; i < 16; ++i) ra[i] = pa[i];
        #pragma unroll
        for (int i = 0; i < 8;  ++i) rb[i] = pb[i];
    };

    auto storeStage = [&]() {
        _Float16* da = &lA[t * LDH];
        _Float16* db = &lB[rowB * LDH + colB];
        #pragma unroll
        for (int i = 0; i < 8; ++i) {
            v8h ha;
            #pragma unroll
            for (int e = 0; e < 4; ++e) {
                ha[e]     = (_Float16)ra[2*i][e];
                ha[4 + e] = (_Float16)ra[2*i + 1][e];
            }
            *(v8h*)(da + i * 8) = ha;
        }
        #pragma unroll
        for (int i = 0; i < 4; ++i) {
            v8h hb;
            #pragma unroll
            for (int e = 0; e < 4; ++e) {
                hb[e]     = (_Float16)(float)rb[2*i][e];     // int8-range: exact in f16
                hb[4 + e] = (_Float16)(float)rb[2*i + 1][e];
            }
            *(v8h*)(db + i * 8) = hb;
        }
    };

    auto compute = [&]() {
        #pragma unroll
        for (int ks = 0; ks < 2; ++ks) {          // two 16x16x32 k-steps per stage
            v16h afrag[4];
            v16h bfrag[4];
            // A fragments: 16-bit A 16x32 layout: g=0 K {0..7,16..23}, g=1 K {8..15,24..31}
            #pragma unroll
            for (int i = 0; i < 4; ++i) {
                const _Float16* p = &lA[(wm * 64 + i * 16 + lr) * LDH + ks * 32];
                v8h lo = *(const v8h*)(p + g * 8);
                v8h hi = *(const v8h*)(p + 16 + g * 8);
                #pragma unroll
                for (int e = 0; e < 8; ++e) { afrag[i][e] = lo[e]; afrag[i][8 + e] = hi[e]; }
            }
            // B fragments: 16-bit B 32x16 layout: lane n = lane%16, g=0 K {0..7,16..23}, g=1 K {8..15,24..31}
            #pragma unroll
            for (int j = 0; j < 4; ++j) {
                const _Float16* p = &lB[(wn * 64 + j * 16 + lr) * LDH + ks * 32 + g * 8];
                v8h lo = *(const v8h*)(p);
                v8h hi = *(const v8h*)(p + 16);
                #pragma unroll
                for (int e = 0; e < 8; ++e) { bfrag[j][e] = lo[e]; bfrag[j][8 + e] = hi[e]; }
            }
            #pragma unroll
            for (int i = 0; i < 4; ++i)
                #pragma unroll
                for (int j = 0; j < 4; ++j)
                    acc[i][j] = __builtin_amdgcn_wmma_f32_16x16x32_f16(
                        false, afrag[i], false, bfrag[j],
                        (short)0, acc[i][j], false, false);
        }
    };

    // Software pipeline: global prefetch of stage kt+1 overlaps WMMA on stage kt.
    loadStage(0);
    for (int kt = 0; kt < NSTAGE; ++kt) {
        __syncthreads();                 // previous compute done with LDS
        storeStage();
        __syncthreads();                 // tile visible to all waves
        if (kt + 1 < NSTAGE) loadStage(kt + 1);
        compute();
    }

    // Epilogue: per-out-channel dequant scale, then store.
    // C/D layout: lane n = lane%16; VGPR r -> row m = r + 8*g (within 16x16 tile).
    float sc[4];
    #pragma unroll
    for (int j = 0; j < 4; ++j)
        sc[j] = SC[n0 + wn * 64 + j * 16 + lr];

    #pragma unroll
    for (int i = 0; i < 4; ++i) {
        #pragma unroll
        for (int j = 0; j < 4; ++j) {
            const int n     = n0 + wn * 64 + j * 16 + lr;
            const int mBase = m0 + wm * 64 + i * 16 + 8 * g;
            #pragma unroll
            for (int r = 0; r < 8; ++r)
                O[(size_t)(mBase + r) * N_TOT + n] = acc[i][j][r] * sc[j];
        }
    }
}

extern "C" void kernel_launch(void* const* d_in, const int* in_sizes, int n_in,
                              void* d_out, int out_size, void* d_ws, size_t ws_size,
                              hipStream_t stream) {
    (void)in_sizes; (void)n_in; (void)out_size; (void)d_ws; (void)ws_size;
    const float* x  = (const float*)d_in[0];
    const int*   w  = (const int*)d_in[1];
    const float* sc = (const float*)d_in[2];
    float*       o  = (float*)d_out;

    dim3 grid(M_TOT / BLK_M, N_TOT / BLK_N);   // (16, 128): M-fast => weight L2 reuse
    dim3 block(256);
    qlin_wmma_f16<<<grid, block, 0, stream>>>(x, w, sc, o);
}
